// MultiHeadAttention_89558658056549
// MI455X (gfx1250) — matrix operand order repaired and checked by reading, not compiled, not run on hardware
//
#include <hip/hip_runtime.h>
#include <hip/hip_bf16.h>

// ---------------------------------------------------------------------------
// MHA for MI455X (gfx1250, wave32, WMMA).
//   batch=8, seq=1024, embed=1024, heads=16, d_k=64
// Pipeline:
//   (1) cast fp32 inputs/weights -> bf16 workspace
//   (2) Q/K/V projections: bf16 WMMA GEMM 8192x1024x1024. Weight panel is
//       staged in LDS with GLOBAL_LOAD_ASYNC_TO_LDS_B128 (ASYNCcnt-tracked,
//       double-buffered) and shared by all 8 waves. Epilogue writes
//       head-split layouts (Q,K: [b,h,s,d]; V transposed: [b,h,d,s]).
//   (3) attention: one block per (b,h,64-query chunk); full 64x1024 fp32
//       score panel lives in 256KB of the WGP's 320KB LDS; exact softmax +
//       the reference's min(p, 1-masked) quirk; PV via WMMA with P fragments
//       rebuilt from LDS on the fly.
//   (4) output projection: same WMMA GEMM, fp32 result + bias -> d_out
// ---------------------------------------------------------------------------

typedef __attribute__((ext_vector_type(16))) __bf16 v16bf;  // 8 VGPRs
typedef __attribute__((ext_vector_type(8)))  float  v8f;    // 8 VGPRs
typedef __attribute__((ext_vector_type(4)))  int    i4x;    // 16B trivial vec
typedef __attribute__((ext_vector_type(4)))  float  f4x;

// address-space-qualified pointer types for the async-LDS builtin
typedef __attribute__((address_space(1))) i4x* gas_i4x_p;   // global
typedef __attribute__((address_space(3))) i4x* las_i4x_p;   // LDS

#define BATCH 8
#define SEQ   1024
#define EMBED 1024
#define HEADS 16
#define DK    64
#define NEGV  (-1000000000.0f)

union FragBF {           // one WMMA bf16 operand (16x32 A or 32x16 B)
    v16bf v;
    i4x   q[2];
};

// ---------------------------------------------------------------------------
// CDNA5 async global->LDS copy (ASYNCcnt) with portable fallback
// ---------------------------------------------------------------------------
#if __has_builtin(__builtin_amdgcn_global_load_async_to_lds_b128)
#define HAS_ASYNC_LDS 1
#else
#define HAS_ASYNC_LDS 0
#endif

__device__ __forceinline__ void async_copy16(const __bf16* g, __bf16* l)
{
#if HAS_ASYNC_LDS
    __builtin_amdgcn_global_load_async_to_lds_b128(
        (gas_i4x_p)(g), (las_i4x_p)(l), 0, 0);
#else
    *(i4x*)l = *(const i4x*)g;
#endif
}

__device__ __forceinline__ void wait_async_le1()
{
#if HAS_ASYNC_LDS
#if __has_builtin(__builtin_amdgcn_s_wait_asynccnt)
    __builtin_amdgcn_s_wait_asynccnt(1);
#else
    asm volatile("s_wait_asynccnt 0x1" ::: "memory");
#endif
#endif
}

__device__ __forceinline__ void wait_async_le0()
{
#if HAS_ASYNC_LDS
#if __has_builtin(__builtin_amdgcn_s_wait_asynccnt)
    __builtin_amdgcn_s_wait_asynccnt(0);
#else
    asm volatile("s_wait_asynccnt 0x0" ::: "memory");
#endif
#endif
}

// ---------------------------------------------------------------------------
// (1) fp32 -> bf16 cast
// ---------------------------------------------------------------------------
__global__ __launch_bounds__(256) void cast_f32_bf16(
    const float* __restrict__ in, __bf16* __restrict__ out, int n)
{
    int i = (blockIdx.x * 256 + threadIdx.x) * 4;
    if (i + 3 < n) {
        f4x v = *(const f4x*)(in + i);
        out[i + 0] = (__bf16)v.x;
        out[i + 1] = (__bf16)v.y;
        out[i + 2] = (__bf16)v.z;
        out[i + 3] = (__bf16)v.w;
    } else {
        for (; i < n; ++i) out[i] = (__bf16)in[i];
    }
}

// ---------------------------------------------------------------------------
// (2)/(4) projection GEMM:  Y[8192,1024] = X @ W^T + bias
//   MODE 0: write bf16, head-split [b,h,s,d]      (Q,K projections)
//   MODE 1: write bf16, head-split [b,h,d,s]      (V projection, transposed)
//   MODE 2: write fp32 row-major [m,n]            (output projection)
// Block: 256 threads = 8 waves. Wave computes a 16x64 strip; block = 128x64.
// B (weight) panel chunk (64 cols x 32 K = 4KB) is async-staged into LDS,
// double-buffered: one b128 async load per thread per chunk.
// Grid: (EMBED/64, 8192/128) = (16, 64).
// ---------------------------------------------------------------------------
template <int MODE>
__global__ __launch_bounds__(256) void proj_kernel(
    const __bf16* __restrict__ X, const __bf16* __restrict__ W,
    const float* __restrict__ bias, __bf16* __restrict__ outb,
    float* __restrict__ outf)
{
    __shared__ __bf16 bbuf[2][64][32];   // double-buffered weight chunk (8KB)

    const int lane  = threadIdx.x & 31;
    const int wave  = threadIdx.x >> 5;
    const int n0    = blockIdx.x * 64;
    const int m0    = blockIdx.y * 128 + wave * 16;

    const int mrow  = lane & 15;         // A: row within tile
    const int khalf = (lane >> 4) * 8;   // unused; A loads use kb
    const int kb    = (lane >> 4) * 16;  // B: K half offset
    const int ncol  = lane & 15;         // B/D: column within tile

    // staging assignment: thread -> (row of 64-col panel, 16B sub-chunk)
    const int sn = threadIdx.x >> 2;         // 0..63  (panel row)
    const int sp = (threadIdx.x & 3) * 8;    // 0,8,16,24 (element offset)
    const __bf16* wrow = W + (size_t)(n0 + sn) * EMBED + sp;

    v8f acc[4];
#pragma unroll
    for (int t = 0; t < 4; ++t)
#pragma unroll
        for (int r = 0; r < 8; ++r) acc[t][r] = 0.0f;

    const __bf16* arow = X + (size_t)(m0 + mrow) * EMBED;

    // prologue: stage chunk 0
    async_copy16(wrow, &bbuf[0][sn][sp]);

    for (int c = 0; c < EMBED / 32; ++c) {
        if (c + 1 < EMBED / 32) {
            async_copy16(wrow + (c + 1) * 32, &bbuf[(c + 1) & 1][sn][sp]);
            wait_async_le1();                  // chunk c has landed
        } else {
            wait_async_le0();
        }
        __syncthreads();                       // all threads' stages visible

        const int k = c * 32;
        if (k + 256 < EMBED)                   // pull A stream toward the WGP
            __builtin_prefetch((const void*)(arow + k + 256), 0, 0);

        FragBF a;   // A[i] = X[m0+mrow][k + kb + i]
        a.q[0] = *(const i4x*)(arow + k + kb);
        a.q[1] = *(const i4x*)(arow + k + kb + 8);
#pragma unroll
        for (int t = 0; t < 4; ++t) {
            FragBF b;   // B[i] = W[n0+t*16+ncol][k + kb + i]  (= W^T column)
            const __bf16* bp = &bbuf[c & 1][t * 16 + ncol][kb];
            b.q[0] = ((const i4x*)bp)[0];
            b.q[1] = ((const i4x*)bp)[1];
            acc[t] = __builtin_amdgcn_wmma_f32_16x16x32_bf16(
                false, a.v, false, b.v, (short)0, acc[t], false, false);
        }
        __syncthreads();                       // done reading buf[c&1]
    }

#pragma unroll
    for (int t = 0; t < 4; ++t) {
        const int   col = n0 + t * 16 + ncol;
        const float bv  = bias[col];
        const int   h   = col >> 6;
        const int   dd  = col & 63;
#pragma unroll
        for (int r = 0; r < 8; ++r) {
            const int   m   = m0 + (lane >> 4) * 8 + r;
            const float val = acc[t][r] + bv;
            const int   bi  = m >> 10;     // batch
            const int   sq  = m & 1023;    // seq position
            if (MODE == 0) {
                outb[(((size_t)(bi * HEADS + h)) * SEQ + sq) * DK + dd] = (__bf16)val;
            } else if (MODE == 1) {
                outb[(((size_t)(bi * HEADS + h)) * DK + dd) * SEQ + sq] = (__bf16)val;
            } else {
                outf[(size_t)m * EMBED + col] = val;
            }
        }
    }
}

// ---------------------------------------------------------------------------
// (3) attention: block = (64 query rows) x (one b,h). 256 threads = 8 waves.
// LDS: S[64][1024] fp32 (256KB) + softmax stats. Uses CDNA5's 320KB LDS.
// ---------------------------------------------------------------------------
__global__ __launch_bounds__(256) void attn_kernel(
    const __bf16* __restrict__ qh, const __bf16* __restrict__ kh,
    const __bf16* __restrict__ vT, const int* __restrict__ cmask,
    __bf16* __restrict__ concat)
{
    extern __shared__ char smem[];
    float* S      = (float*)smem;                            // [64][1024]
    float* pmax   = (float*)(smem + (size_t)64 * 1024 * 4);  // [64][4]
    float* psum   = pmax + 256;                              // [64][4]
    float* rowMax = psum + 256;                              // [64]
    float* rowSum = rowMax + 64;                             // [64]

    const int tid  = threadIdx.x;
    const int lane = tid & 31;
    const int wave = tid >> 5;
    const int qc = blockIdx.x;      // query chunk 0..15
    const int h  = blockIdx.y;
    const int b  = blockIdx.z;
    const int q0 = qc * 64;

    const size_t headoff = (size_t)(b * HEADS + h) * SEQ * DK;
    const __bf16* Q = qh + headoff;      // [SEQ][DK]
    const __bf16* K = kh + headoff;      // [SEQ][DK]
    const __bf16* V = vT + headoff;      // [DK][SEQ] (transposed)
    const int*    cm = cmask + b * SEQ;

    const int mrow  = lane & 15;
    const int khalf = (lane >> 4) * 8;
    const int kb    = (lane >> 4) * 16;
    const int ncol  = lane & 15;

    // ---- Stage A: S = mask(scale(Q @ K^T)) into LDS -----------------------
    {
        const int mt     = wave >> 1;            // 0..3 (16-row strip)
        const int ntbase = (wave & 1) * 32;      // 32 key tiles per wave

        FragBF qa[2];
        const __bf16* qrow = Q + (size_t)(q0 + mt * 16 + mrow) * DK;
#pragma unroll
        for (int ks = 0; ks < 2; ++ks) {
            qa[ks].q[0] = *(const i4x*)(qrow + ks * 32 + kb);
            qa[ks].q[1] = *(const i4x*)(qrow + ks * 32 + kb + 8);
        }

        for (int nt = ntbase; nt < ntbase + 32; ++nt) {
            v8f acc;
#pragma unroll
            for (int r = 0; r < 8; ++r) acc[r] = 0.0f;

            const __bf16* krow = K + (size_t)(nt * 16 + ncol) * DK + kb;
#pragma unroll
            for (int ks = 0; ks < 2; ++ks) {
                FragBF kf;   // B[i] = K[key][kd]  -> K^T fragment
                kf.q[0] = ((const i4x*)(krow + ks * 32))[0];
                kf.q[1] = ((const i4x*)(krow + ks * 32))[1];
                acc = __builtin_amdgcn_wmma_f32_16x16x32_bf16(
                    false, qa[ks].v, false, kf.v, (short)0, acc, false, false);
            }

            const int  key    = nt * 16 + ncol;
            const bool masked = (cm[key] == 1);
#pragma unroll
            for (int r = 0; r < 8; ++r) {
                float s = acc[r] * 0.125f;           // 1/sqrt(64)
                s = masked ? NEGV : s;
                const int row = mt * 16 + (lane >> 4) * 8 + r;   // 0..63
                S[row * 1024 + key] = s;
            }
        }
    }
    __syncthreads();

    // ---- Stage B: exact softmax statistics (two-pass over LDS panel) ------
    {
        const int row = tid & 63;
        const int qtr = tid >> 6;                 // 0..3
        const float* srow = S + row * 1024 + qtr * 256;

        float mx = -3.4e38f;
        for (int i = 0; i < 256; ++i) mx = fmaxf(mx, srow[i]);
        pmax[row * 4 + qtr] = mx;
        __syncthreads();
        if (tid < 64) {
            rowMax[tid] = fmaxf(fmaxf(pmax[tid * 4 + 0], pmax[tid * 4 + 1]),
                                fmaxf(pmax[tid * 4 + 2], pmax[tid * 4 + 3]));
        }
        __syncthreads();

        const float rm = rowMax[row];
        float sum = 0.0f;
        for (int i = 0; i < 256; ++i) sum += __expf(srow[i] - rm);
        psum[row * 4 + qtr] = sum;
        __syncthreads();
        if (tid < 64) {
            rowSum[tid] = psum[tid * 4 + 0] + psum[tid * 4 + 1] +
                          psum[tid * 4 + 2] + psum[tid * 4 + 3];
        }
        __syncthreads();
    }

    // ---- Stage C: O = P @ V; P rebuilt from LDS with exp + quirky min -----
    {
        const int mt = wave >> 1;                 // 0..3
        const int nh = (wave & 1) * 32;           // output-column half
        const int prow = mt * 16 + mrow;          // this lane's P row
        const float rm = rowMax[prow];
        const float ri = 1.0f / rowSum[prow];

        v8f acc[2];
#pragma unroll
        for (int t = 0; t < 2; ++t)
#pragma unroll
            for (int r = 0; r < 8; ++r) acc[t][r] = 0.0f;

        for (int k = 0; k < SEQ; k += 32) {
            FragBF pf;   // A[i] = P[prow][k + kb + i]
            const float* sp = S + prow * 1024 + k + kb;
#pragma unroll
            for (int i = 0; i < 8; ++i) {
                float s = sp[i];
                float p = fminf(__expf(s - rm) * ri, 1.0f - s);
                pf.v[i] = (__bf16)p;
            }
#pragma unroll
            for (int i = 0; i < 8; ++i) {
                float s = sp[8 + i];
                float p = fminf(__expf(s - rm) * ri, 1.0f - s);
                pf.v[8 + i] = (__bf16)p;
            }
#pragma unroll
            for (int t = 0; t < 2; ++t) {
                FragBF vf;   // B[i] = V[key = k+kb+i][dd] = vT[dd][key]
                const __bf16* vp = V + (size_t)(nh + t * 16 + ncol) * SEQ + k + kb;
                vf.q[0] = ((const i4x*)vp)[0];
                vf.q[1] = ((const i4x*)vp)[1];
                acc[t] = __builtin_amdgcn_wmma_f32_16x16x32_bf16(
                    false, pf.v, false, vf.v, (short)0, acc[t], false, false);
            }
        }

        // write concat [b, sq, h*64+dd] as bf16 for the output projection
#pragma unroll
        for (int t = 0; t < 2; ++t) {
            const int dd = nh + t * 16 + ncol;
#pragma unroll
            for (int r = 0; r < 8; ++r) {
                const int sq = q0 + mt * 16 + (lane >> 4) * 8 + r;
                concat[((size_t)b * SEQ + sq) * EMBED + h * DK + dd] =
                    (__bf16)acc[t][r];
            }
        }
    }
}

// ---------------------------------------------------------------------------
// host launcher
// ---------------------------------------------------------------------------
extern "C" void kernel_launch(void* const* d_in, const int* in_sizes, int n_in,
                              void* d_out, int out_size, void* d_ws, size_t ws_size,
                              hipStream_t stream) {
    const float* q   = (const float*)d_in[0];
    const float* k   = (const float*)d_in[1];
    const float* v   = (const float*)d_in[2];
    const int*   cm  = (const int*)d_in[3];
    const float* Wq  = (const float*)d_in[4];
    const float* bq  = (const float*)d_in[5];
    const float* Wk  = (const float*)d_in[6];
    const float* bk  = (const float*)d_in[7];
    const float* Wv  = (const float*)d_in[8];
    const float* bv  = (const float*)d_in[9];
    const float* Wo  = (const float*)d_in[10];
    const float* bo  = (const float*)d_in[11];

    char* ws = (char*)d_ws;
    const size_t XB = (size_t)BATCH * SEQ * EMBED * 2;  // 16 MB bf16 tensor
    const size_t WB = (size_t)EMBED * EMBED * 2;        //  2 MB bf16 weight

    __bf16* xqb = (__bf16*)(ws + 0 * XB);
    __bf16* xkb = (__bf16*)(ws + 1 * XB);
    __bf16* xvb = (__bf16*)(ws + 2 * XB);
    __bf16* wqb = (__bf16*)(ws + 3 * XB + 0 * WB);
    __bf16* wkb = (__bf16*)(ws + 3 * XB + 1 * WB);
    __bf16* wvb = (__bf16*)(ws + 3 * XB + 2 * WB);
    __bf16* wob = (__bf16*)(ws + 3 * XB + 3 * WB);
    __bf16* qhp = (__bf16*)(ws + 3 * XB + 4 * WB);
    __bf16* khp = (__bf16*)(ws + 4 * XB + 4 * WB);
    __bf16* vTp = (__bf16*)(ws + 5 * XB + 4 * WB);
    __bf16* cat = xqb;   // xq no longer needed once projections are done

    const int nx = BATCH * SEQ * EMBED;   // 8388608
    const int nw = EMBED * EMBED;         // 1048576

    cast_f32_bf16<<<nx / 1024, 256, 0, stream>>>(q,  xqb, nx);
    cast_f32_bf16<<<nx / 1024, 256, 0, stream>>>(k,  xkb, nx);
    cast_f32_bf16<<<nx / 1024, 256, 0, stream>>>(v,  xvb, nx);
    cast_f32_bf16<<<nw / 1024, 256, 0, stream>>>(Wq, wqb, nw);
    cast_f32_bf16<<<nw / 1024, 256, 0, stream>>>(Wk, wkb, nw);
    cast_f32_bf16<<<nw / 1024, 256, 0, stream>>>(Wv, wvb, nw);
    cast_f32_bf16<<<nw / 1024, 256, 0, stream>>>(Wo, wob, nw);

    dim3 gproj(EMBED / 64, (BATCH * SEQ) / 128);   // (16, 64)
    proj_kernel<0><<<gproj, 256, 0, stream>>>(xqb, wqb, bq, qhp, nullptr);
    proj_kernel<0><<<gproj, 256, 0, stream>>>(xkb, wkb, bk, khp, nullptr);
    proj_kernel<1><<<gproj, 256, 0, stream>>>(xvb, wvb, bv, vTp, nullptr);

    const size_t smem = (size_t)64 * 1024 * 4 + (256 + 256 + 64 + 64) * 4;
    attn_kernel<<<dim3(SEQ / 64, HEADS, BATCH), 256, smem, stream>>>(
        qhp, khp, vTp, cm, cat);

    proj_kernel<2><<<gproj, 256, 0, stream>>>(cat, wob, bo, nullptr,
                                              (float*)d_out);
}
